// GIN_Graph_34497177322039
// MI455X (gfx1250) — matrix operand order repaired and checked by reading, not compiled, not run on hardware
//
#include <hip/hip_runtime.h>
#include <hip/hip_bf16.h>

#define FDIM 128

typedef __attribute__((ext_vector_type(16))) _Float16 v16h;
typedef __attribute__((ext_vector_type(8)))  _Float16 v8h;
typedef __attribute__((ext_vector_type(8)))  float    v8f;

// ---------------------------------------------------------------- helpers

__device__ __forceinline__ v8f wmma_hilo(v16h aH, v16h aL, v16h bH, v16h bL, v8f c) {
    // D = A*B + C with ~f32 accuracy: hi*hi + hi*lo + lo*hi (f32 accumulate)
    c = __builtin_amdgcn_wmma_f32_16x16x32_f16(false, aH, false, bH, (short)0, c, false, false);
    c = __builtin_amdgcn_wmma_f32_16x16x32_f16(false, aH, false, bL, (short)0, c, false, false);
    c = __builtin_amdgcn_wmma_f32_16x16x32_f16(false, aL, false, bH, (short)0, c, false, false);
    return c;
}

// A-fragment (16-bit, 16x32 MxK layout): lane m=lane&15, hid=lane>>4.
// Elements e=0..7  -> K = off0+e,  e=8..15 -> K = off0+16+(e-8)
// with off0 = kt*32 + hid*8  (per CDNA5 ISA 7.12.2 table).
__device__ __forceinline__ v16h load_a_frag(const _Float16* row, int off0) {
    v8h a = *(const v8h*)(row + off0);
    v8h b = *(const v8h*)(row + off0 + 16);
    return __builtin_shufflevector(a, b, 0,1,2,3,4,5,6,7,8,9,10,11,12,13,14,15);
}

__device__ __forceinline__ void split_store(_Float16* hi, _Float16* lo, int idx, float v) {
    _Float16 h = (_Float16)v;
    hi[idx] = h;
    lo[idx] = (_Float16)(v - (float)h);
}

__device__ __forceinline__ void atomicMaxFloat(float* addr, float v) {
    if (v >= 0.0f) atomicMax((int*)addr, __float_as_int(v));
    else           atomicMin((unsigned int*)addr, __float_as_uint(v));
}

// ---------------------------------------------------------------- small kernels

__global__ void fill_f32(float* __restrict__ p, float v, int n) {
    int i = blockIdx.x * blockDim.x + threadIdx.x;
    if (i < n) p[i] = v;
}

// W [K][N] f32 (row-major) -> WT hi/lo halves [N][K]
__global__ void convert_w(const float* __restrict__ W, _Float16* __restrict__ WTh,
                          _Float16* __restrict__ WTl, int K, int N) {
    int i = blockIdx.x * blockDim.x + threadIdx.x;
    if (i >= K * N) return;
    int k = i / N, n = i - k * N;
    float w = W[i];
    _Float16 h = (_Float16)w;
    WTh[(size_t)n * K + k] = h;
    WTl[(size_t)n * K + k] = (_Float16)(w - (float)h);
}

// agg[i] = (1 + eps[l]) * x[i]
__global__ void init_agg(const float* __restrict__ X, const float* __restrict__ eps, int l,
                         float* __restrict__ A, int n4) {
    int i = blockIdx.x * blockDim.x + threadIdx.x;
    if (i >= n4) return;
    float e = 1.0f + eps[l];
    float4 v = ((const float4*)X)[i];
    v.x *= e; v.y *= e; v.z *= e; v.w *= e;
    ((float4*)A)[i] = v;
}

// agg[dst] += x[src], 32 lanes per edge (4 floats each)
__global__ void edge_scatter(const float* __restrict__ X, const int* __restrict__ ei,
                             int nE, float* __restrict__ A) {
    long long t = (long long)blockIdx.x * blockDim.x + threadIdx.x;
    int e = (int)(t >> 5);
    if (e >= nE) return;
    int q = (int)(t & 31);
    int s = ei[e];
    int d = ei[(size_t)nE + e];
    float4 v = ((const float4*)(X + (size_t)s * FDIM))[q];
    float* ap = A + (size_t)d * FDIM + q * 4;
    atomicAdd(ap + 0, v.x);
    atomicAdd(ap + 1, v.y);
    atomicAdd(ap + 2, v.z);
    atomicAdd(ap + 3, v.w);
}

// column sums / sums of squares over M rows
__global__ void bn_stats(const float* __restrict__ H, int M, float* __restrict__ sums) {
    int tid = threadIdx.x;                 // 256 threads
    int col = tid & 127, sub = tid >> 7;
    int r = blockIdx.x * 512 + sub;
    int rEnd = blockIdx.x * 512 + 512;
    if (rEnd > M) rEnd = M;
    float s = 0.0f, q = 0.0f;
    for (; r < rEnd; r += 2) {
        float v = H[(size_t)r * FDIM + col];
        s += v; q += v * v;
    }
    atomicAdd(&sums[col], s);
    atomicAdd(&sums[FDIM + col], q);
}

__global__ void bn_finalize(const float* __restrict__ sums, const float* __restrict__ g,
                            const float* __restrict__ b, float invM,
                            float* __restrict__ scale, float* __restrict__ shift) {
    int c = threadIdx.x;                   // 128 threads
    float mean = sums[c] * invM;
    float var  = sums[FDIM + c] * invM - mean * mean;
    float rstd = rsqrtf(var + 1e-5f);
    float sc = g[c] * rstd;
    scale[c] = sc;
    shift[c] = b[c] - mean * sc;
}

// per-node pooling: gmax/gsum/cnt by graph id
__global__ void pool_scatter(const float* __restrict__ X, const int* __restrict__ batch, int nN,
                             float* __restrict__ gmax, float* __restrict__ gsum,
                             float* __restrict__ cnt) {
    long long t = (long long)blockIdx.x * blockDim.x + threadIdx.x;
    int node = (int)(t >> 5);
    if (node >= nN) return;
    int q = (int)(t & 31);
    int g = batch[node];
    float4 v = ((const float4*)(X + (size_t)node * FDIM))[q];
    float* mp = gmax + (size_t)g * FDIM + q * 4;
    float* sp = gsum + (size_t)g * FDIM + q * 4;
    atomicMaxFloat(mp + 0, v.x); atomicMaxFloat(mp + 1, v.y);
    atomicMaxFloat(mp + 2, v.z); atomicMaxFloat(mp + 3, v.w);
    atomicAdd(sp + 0, v.x); atomicAdd(sp + 1, v.y);
    atomicAdd(sp + 2, v.z); atomicAdd(sp + 3, v.w);
    if (q == 0) atomicAdd(&cnt[g], 1.0f);
}

// ---------------------------------------------------------------- WMMA node GEMM
// One wave (32 threads) per 16-row tile.  Out[M x 128] = act(A[M x 128]) @ W[128 x 128] + bias
// PRE:  a' = relu(a*preScale[c] + preShift[c])   (fused BN+ReLU for GEMM2 input)
// LN :  row LayerNorm + leaky_relu(0.1) epilogue (GEMM2 output)
template <bool PRE, bool LN>
__global__ void __launch_bounds__(32)
gemm_node(const float* __restrict__ A,
          const _Float16* __restrict__ WTh, const _Float16* __restrict__ WTl,
          const float* __restrict__ bias,
          const float* __restrict__ preScale, const float* __restrict__ preShift,
          const float* __restrict__ lnG, const float* __restrict__ lnB,
          float* __restrict__ Out, int M) {
    __shared__ __align__(32) _Float16 shHi[16 * FDIM];
    __shared__ __align__(32) _Float16 shLo[16 * FDIM];
    __shared__ float shC[16 * FDIM];
    __shared__ float shMean[16], shRstd[16];

    const int lane = threadIdx.x;
    const int row0 = blockIdx.x * 16;

    // stage A tile into LDS (hi/lo split, optional fused BN-affine + ReLU)
    for (int i = lane; i < 16 * FDIM / 4; i += 32) {
        int idx = i * 4;
        int r = idx >> 7, c = idx & 127;
        int rr = row0 + r; if (rr >= M) rr = M - 1;
        float4 v = *(const float4*)(A + (size_t)rr * FDIM + c);
        if (PRE) {
            v.x = fmaxf(fmaf(v.x, preScale[c + 0], preShift[c + 0]), 0.0f);
            v.y = fmaxf(fmaf(v.y, preScale[c + 1], preShift[c + 1]), 0.0f);
            v.z = fmaxf(fmaf(v.z, preScale[c + 2], preShift[c + 2]), 0.0f);
            v.w = fmaxf(fmaf(v.w, preScale[c + 3], preShift[c + 3]), 0.0f);
        }
        split_store(shHi, shLo, idx + 0, v.x);
        split_store(shHi, shLo, idx + 1, v.y);
        split_store(shHi, shLo, idx + 2, v.z);
        split_store(shHi, shLo, idx + 3, v.w);
    }
    __syncthreads();

    const int m = lane & 15, hid = lane >> 4;
    const _Float16* rowHi = shHi + m * FDIM;
    const _Float16* rowLo = shLo + m * FDIM;

    for (int nt = 0; nt < 8; ++nt) {
        const int ncol = nt * 16 + m;
        const _Float16* bHp = WTh + (size_t)ncol * FDIM;
        const _Float16* bLp = WTl + (size_t)ncol * FDIM;
        v8f acc = {0.f, 0.f, 0.f, 0.f, 0.f, 0.f, 0.f, 0.f};
#pragma unroll
        for (int kt = 0; kt < 4; ++kt) {
            int off0 = kt * 32 + hid * 8;
            v16h aH = load_a_frag(rowHi, off0);
            v16h aL = load_a_frag(rowLo, off0);
            int kb = kt * 32 + hid * 8;        // B: K = kb + e (e<8), kb + 16 + (e-8) (e>=8), N = lane&15
            v16h bH = load_a_frag(bHp, kb);
            v16h bL = load_a_frag(bLp, kb);
            acc = wmma_hilo(aH, aL, bH, bL, acc);
        }
        float bb = bias[ncol];
        if (!LN) {
#pragma unroll
            for (int r = 0; r < 8; ++r) {
                int gr = row0 + r + hid * 8;   // C/D layout: lanes 0-15 M=r, 16-31 M=r+8
                if (gr < M) Out[(size_t)gr * FDIM + ncol] = acc[r] + bb;
            }
        } else {
#pragma unroll
            for (int r = 0; r < 8; ++r)
                shC[(r + hid * 8) * FDIM + ncol] = acc[r] + bb;
        }
    }

    if (LN) {
        __syncthreads();
        int row = lane >> 1, seg = lane & 1, cb = seg * 64;
        float s = 0.0f, q = 0.0f;
#pragma unroll
        for (int j = 0; j < 16; ++j) {
            float4 v = *(const float4*)(shC + row * FDIM + cb + j * 4);
            s += v.x + v.y + v.z + v.w;
            q += v.x * v.x + v.y * v.y + v.z * v.z + v.w * v.w;
        }
        s += __shfl_xor(s, 1, 32);
        q += __shfl_xor(q, 1, 32);
        float mean = s * (1.0f / 128.0f);
        float var  = q * (1.0f / 128.0f) - mean * mean;
        shMean[row] = mean;
        shRstd[row] = rsqrtf(var + 1e-5f);
        __syncthreads();
        for (int i = lane; i < 16 * FDIM / 4; i += 32) {
            int idx = i * 4;
            int r = idx >> 7, c = idx & 127;
            int gr = row0 + r;
            float mu = shMean[r], rs = shRstd[r];
            float4 v = *(const float4*)(shC + idx);
            float4 o;
            o.x = (v.x - mu) * rs * lnG[c + 0] + lnB[c + 0];
            o.y = (v.y - mu) * rs * lnG[c + 1] + lnB[c + 1];
            o.z = (v.z - mu) * rs * lnG[c + 2] + lnB[c + 2];
            o.w = (v.w - mu) * rs * lnG[c + 3] + lnB[c + 3];
            o.x = o.x >= 0.0f ? o.x : 0.1f * o.x;
            o.y = o.y >= 0.0f ? o.y : 0.1f * o.y;
            o.z = o.z >= 0.0f ? o.z : 0.1f * o.z;
            o.w = o.w >= 0.0f ? o.w : 0.1f * o.w;
            if (gr < M) *(float4*)(Out + (size_t)gr * FDIM + c) = o;
        }
    }
}

// ---------------------------------------------------------------- final pooled GEMM
// Out[G x 128] = [gmax || gsum/cnt][G x 256] @ Wout[256 x 128] + bout
__global__ void __launch_bounds__(32)
final_gemm(const float* __restrict__ gmax, const float* __restrict__ gsum,
           const float* __restrict__ cnt,
           const _Float16* __restrict__ WTh, const _Float16* __restrict__ WTl,
           const float* __restrict__ bout, float* __restrict__ Out, int G) {
    __shared__ __align__(32) _Float16 shHi[16 * 256];
    __shared__ __align__(32) _Float16 shLo[16 * 256];
    const int lane = threadIdx.x;
    const int g0 = blockIdx.x * 16;

    for (int i = lane; i < 16 * 256 / 4; i += 32) {
        int idx = i * 4;
        int r = idx >> 8, c = idx & 255;
        int g = g0 + r; if (g >= G) g = G - 1;
        float4 v;
        if (c < 128) {
            v = *(const float4*)(gmax + (size_t)g * FDIM + c);
        } else {
            float rc = 1.0f / fmaxf(cnt[g], 1.0f);
            float4 sv = *(const float4*)(gsum + (size_t)g * FDIM + (c - 128));
            v = make_float4(sv.x * rc, sv.y * rc, sv.z * rc, sv.w * rc);
        }
        split_store(shHi, shLo, idx + 0, v.x);
        split_store(shHi, shLo, idx + 1, v.y);
        split_store(shHi, shLo, idx + 2, v.z);
        split_store(shHi, shLo, idx + 3, v.w);
    }
    __syncthreads();

    const int m = lane & 15, hid = lane >> 4;
    const _Float16* rowHi = shHi + m * 256;
    const _Float16* rowLo = shLo + m * 256;

    for (int nt = 0; nt < 8; ++nt) {
        const int ncol = nt * 16 + m;
        const _Float16* bHp = WTh + (size_t)ncol * 256;
        const _Float16* bLp = WTl + (size_t)ncol * 256;
        v8f acc = {0.f, 0.f, 0.f, 0.f, 0.f, 0.f, 0.f, 0.f};
#pragma unroll
        for (int kt = 0; kt < 8; ++kt) {
            int off0 = kt * 32 + hid * 8;
            v16h aH = load_a_frag(rowHi, off0);
            v16h aL = load_a_frag(rowLo, off0);
            int kb = kt * 32 + hid * 8;
            v16h bH = load_a_frag(bHp, kb);
            v16h bL = load_a_frag(bLp, kb);
            acc = wmma_hilo(aH, aL, bH, bL, acc);
        }
        float bb = bout[ncol];
#pragma unroll
        for (int r = 0; r < 8; ++r) {
            int gg = g0 + r + hid * 8;
            if (gg < G) Out[(size_t)gg * FDIM + ncol] = acc[r] + bb;
        }
    }
}

// ---------------------------------------------------------------- launch

extern "C" void kernel_launch(void* const* d_in, const int* in_sizes, int n_in,
                              void* d_out, int out_size, void* d_ws, size_t ws_size,
                              hipStream_t stream) {
    const float* x     = (const float*)d_in[0];
    const float* W1    = (const float*)d_in[1];
    const float* b1    = (const float*)d_in[2];
    const float* bn_g  = (const float*)d_in[3];
    const float* bn_b  = (const float*)d_in[4];
    const float* W2    = (const float*)d_in[5];
    const float* b2    = (const float*)d_in[6];
    const float* geps  = (const float*)d_in[7];
    const float* ln_g  = (const float*)d_in[8];
    const float* ln_b  = (const float*)d_in[9];
    const float* Wout  = (const float*)d_in[10];
    const float* bout  = (const float*)d_in[11];
    const int*   eidx  = (const int*)d_in[12];
    const int*   batch = (const int*)d_in[13];
    (void)n_in; (void)ws_size;

    const int NN = in_sizes[0] / FDIM;
    const int NE = in_sizes[12] / 2;
    const int L  = in_sizes[1] / (FDIM * FDIM);
    const int G  = out_size / FDIM;
    float* out = (float*)d_out;

    // workspace carve
    char* ws = (char*)d_ws;
    size_t nodeBytes = (size_t)NN * FDIM * sizeof(float);
    float* B0 = (float*)ws;
    float* B1 = (float*)(ws + nodeBytes);
    size_t o = 2 * nodeBytes;
    auto carve = [&](size_t bytes) -> void* {
        void* p = ws + o;
        o += (bytes + 255) & ~(size_t)255;
        return p;
    };
    const size_t wHalf = (size_t)FDIM * FDIM * sizeof(_Float16);
    _Float16 *w1h[3], *w1l[3], *w2h[3], *w2l[3];
    for (int l = 0; l < L; ++l) { w1h[l] = (_Float16*)carve(wHalf); w1l[l] = (_Float16*)carve(wHalf); }
    for (int l = 0; l < L; ++l) { w2h[l] = (_Float16*)carve(wHalf); w2l[l] = (_Float16*)carve(wHalf); }
    _Float16* woh = (_Float16*)carve((size_t)2 * FDIM * FDIM * sizeof(_Float16));
    _Float16* wol = (_Float16*)carve((size_t)2 * FDIM * FDIM * sizeof(_Float16));
    float* stats = (float*)carve(512 * sizeof(float));        // sum|sumsq|scale|shift
    float* gmax  = (float*)carve((size_t)G * FDIM * sizeof(float));
    float* gsum  = (float*)carve((size_t)G * FDIM * sizeof(float));
    float* cnt   = (float*)carve((size_t)G * sizeof(float));

    // weight conversion (hi/lo split, transposed to [N][K])
    {
        int nw = FDIM * FDIM, grid = (nw + 255) / 256;
        for (int l = 0; l < L; ++l) {
            convert_w<<<grid, 256, 0, stream>>>(W1 + (size_t)l * nw, w1h[l], w1l[l], FDIM, FDIM);
            convert_w<<<grid, 256, 0, stream>>>(W2 + (size_t)l * nw, w2h[l], w2l[l], FDIM, FDIM);
        }
        int nwo = 2 * FDIM * FDIM;
        convert_w<<<(nwo + 255) / 256, 256, 0, stream>>>(Wout, woh, wol, 2 * FDIM, FDIM);
    }

    const int n4 = NN * FDIM / 4;
    const unsigned gridElem = (unsigned)((n4 + 255) / 256);
    const unsigned gridEdge = (unsigned)(((size_t)NE * 32 + 255) / 256);
    const unsigned gridNode = (unsigned)(((size_t)NN * 32 + 255) / 256);
    const unsigned gridGemm = (unsigned)((NN + 15) / 16);
    const unsigned gridBN   = (unsigned)((NN + 511) / 512);

    const float* prev = x;
    float* bufA = B0;
    float* bufH = B1;
    for (int l = 0; l < L; ++l) {
        float* agg = bufA;
        float* h   = bufH;
        float* nxt = bufA;   // reuse agg buffer for layer output

        init_agg<<<gridElem, 256, 0, stream>>>(prev, geps, l, agg, n4);
        edge_scatter<<<gridEdge, 256, 0, stream>>>(prev, eidx, NE, agg);

        gemm_node<false, false><<<gridGemm, 32, 0, stream>>>(
            agg, w1h[l], w1l[l], b1 + (size_t)l * FDIM,
            nullptr, nullptr, nullptr, nullptr, h, NN);

        fill_f32<<<1, 256, 0, stream>>>(stats, 0.0f, 256);
        bn_stats<<<gridBN, 256, 0, stream>>>(h, NN, stats);
        bn_finalize<<<1, 128, 0, stream>>>(stats, bn_g + (size_t)l * FDIM,
                                           bn_b + (size_t)l * FDIM, 1.0f / (float)NN,
                                           stats + 256, stats + 384);

        gemm_node<true, true><<<gridGemm, 32, 0, stream>>>(
            h, w2h[l], w2l[l], b2 + (size_t)l * FDIM,
            stats + 256, stats + 384,
            ln_g + (size_t)l * FDIM, ln_b + (size_t)l * FDIM, nxt, NN);

        prev = nxt;
        float* t = bufA; bufA = bufH; bufH = t;
    }

    // pooling
    int gp = G * FDIM;
    fill_f32<<<(gp + 255) / 256, 256, 0, stream>>>(gmax, -3.402823466e38f, gp);
    fill_f32<<<(gp + 255) / 256, 256, 0, stream>>>(gsum, 0.0f, gp);
    fill_f32<<<(G + 255) / 256, 256, 0, stream>>>(cnt, 0.0f, G);
    pool_scatter<<<gridNode, 256, 0, stream>>>(prev, batch, NN, gmax, gsum, cnt);

    final_gemm<<<(G + 15) / 16, 32, 0, stream>>>(gmax, gsum, cnt, woh, wol, bout, out, G);
}
